// CausalWanSelfAttention_62491774157261
// MI455X (gfx1250) — matrix operand order repaired and checked by reading, not compiled, not run on hardware
//
#include <hip/hip_runtime.h>
#include <hip/hip_bf16.h>
#include <cmath>

typedef __attribute__((ext_vector_type(16))) _Float16 v16h;
typedef __attribute__((ext_vector_type(8)))  _Float16 v8h;
typedef __attribute__((ext_vector_type(8)))  float    v8f;

#define L_TOK   7040
#define DIMN    1536
#define NHEAD   12
#define HDIM    128
#define KDIM    1536
#define FRAME   880
#define BLOCK_Q 1760
#define SINK    880
#define NQT     (L_TOK / 16)   // 440
#define DT_C    22
#define DS_C    21

// ---------------- WMMA fragment helpers (16x16x32 f16 -> f32) ----------------
// A-matrix 16x32 (MxK): lane m = lane&15, group g = lane>>4.
// halves[i] = K[k0 + 16g + i] (contiguous 32B)
static __device__ __forceinline__ v16h load_fragA(const _Float16* __restrict__ p,
                                                  int ld, int row0, int k0, int lane) {
  int m = lane & 15, g = lane >> 4;
  const _Float16* q = p + (size_t)(row0 + m) * (size_t)ld + k0 + g * 16;
  v8h lo = *(const v8h*)q;
  v8h hi = *(const v8h*)(q + 8);
  v16h r;
#pragma unroll
  for (int i = 0; i < 8; i++) { r[i] = lo[i]; r[8 + i] = hi[i]; }
  return r;
}

// B-matrix 32x16 (KxN), loaded from row-major [N][K] (i.e. B^T in memory):
// lane n = lane&15, group g = lane>>4; halves[i] = mem[n][k0 + 16g + i] (contiguous 32B)
static __device__ __forceinline__ v16h load_fragB(const _Float16* __restrict__ p,
                                                  int ld, int n0, int k0, int lane) {
  int n = lane & 15, g = lane >> 4;
  const _Float16* q = p + (size_t)(n0 + n) * (size_t)ld + k0 + g * 16;
  v8h lo = *(const v8h*)q;
  v8h hi = *(const v8h*)(q + 8);
  v16h r;
#pragma unroll
  for (int i = 0; i < 8; i++) { r[i] = lo[i]; r[8 + i] = hi[i]; }
  return r;
}

static __device__ __forceinline__ v8f wmma_f16(v16h a, v16h b, v8f c) {
  return __builtin_amdgcn_wmma_f32_16x16x32_f16(false, a, false, b, (short)0, c,
                                                false, false);
}

// ---------------- cast fp32 -> f16 ----------------
__global__ void cast_kernel(const float* __restrict__ in, _Float16* __restrict__ out,
                            long n) {
  long i = (long)blockIdx.x * blockDim.x + threadIdx.x;
  if (i < n) out[i] = (_Float16)in[i];
}

// ---------------- GEMM: C[M,N] = A[M,K] * W[N,K]^T + bias[N] (fp32 out) --------
// block = 256 threads = 8 waves arranged 2(M) x 4(N); block tile 128 x 128;
// wave tile 64(M) x 32(N): 8 wmma per 6 fragment loads per k-step.
__global__ __launch_bounds__(256) void gemm_kernel(const _Float16* __restrict__ A,
                                                   const _Float16* __restrict__ W,
                                                   const float* __restrict__ bias,
                                                   float* __restrict__ C,
                                                   int N, int K) {
  int lane = threadIdx.x & 31;
  int wave = threadIdx.x >> 5;
  int row0 = blockIdx.x * 128 + (wave & 1) * 64;
  int col0 = blockIdx.y * 128 + (wave >> 1) * 32;

  v8f c0[4], c1[4];
  v8f zero = {};
#pragma unroll
  for (int i = 0; i < 4; i++) { c0[i] = zero; c1[i] = zero; }

  for (int k = 0; k < K; k += 32) {
    v16h b0 = load_fragB(W, K, col0, k, lane);
    v16h b1 = load_fragB(W, K, col0 + 16, k, lane);
    v16h a0 = load_fragA(A, K, row0, k, lane);
    v16h a1 = load_fragA(A, K, row0 + 16, k, lane);
    v16h a2 = load_fragA(A, K, row0 + 32, k, lane);
    v16h a3 = load_fragA(A, K, row0 + 48, k, lane);
    c0[0] = wmma_f16(a0, b0, c0[0]);
    c1[0] = wmma_f16(a0, b1, c1[0]);
    c0[1] = wmma_f16(a1, b0, c0[1]);
    c1[1] = wmma_f16(a1, b1, c1[1]);
    c0[2] = wmma_f16(a2, b0, c0[2]);
    c1[2] = wmma_f16(a2, b1, c1[2]);
    c0[3] = wmma_f16(a3, b0, c0[3]);
    c1[3] = wmma_f16(a3, b1, c1[3]);
  }
  int n = lane & 15, g = lane >> 4;
  float b0v = bias[col0 + n];
  float b1v = bias[col0 + 16 + n];
#pragma unroll
  for (int i = 0; i < 4; i++) {
#pragma unroll
    for (int r = 0; r < 8; r++) {
      int m0 = row0 + i * 16 + r + 8 * g;
      C[(size_t)m0 * N + col0 + n]      = c0[i][r] + b0v;
      C[(size_t)m0 * N + col0 + 16 + n] = c1[i][r] + b1v;
    }
  }
}

// ---------------- RMSNorm(full 1536) * gain + 3-axis RoPE, write f16 [NH][L][HD]
__global__ __launch_bounds__(256) void rmsrope_kernel(const float* __restrict__ T,
                                                      const float* __restrict__ gain,
                                                      const float* __restrict__ fc,
                                                      const float* __restrict__ fs,
                                                      _Float16* __restrict__ out) {
  __shared__ float red[8];
  int l = blockIdx.x;
  int tid = threadIdx.x;
  const float* row = T + (size_t)l * DIMN;
  int base = tid * 6;
  float v[6];
  float ss = 0.f;
#pragma unroll
  for (int i = 0; i < 6; i++) { v[i] = row[base + i]; ss += v[i] * v[i]; }
#pragma unroll
  for (int off = 16; off; off >>= 1) ss += __shfl_xor(ss, off, 32);
  if ((tid & 31) == 0) red[tid >> 5] = ss;
  __syncthreads();
  float tot = 0.f;
#pragma unroll
  for (int i = 0; i < 8; i++) tot += red[i];
  float rstd = rsqrtf(tot * (1.0f / DIMN) + 1e-6f);

  int f = l / FRAME;
  int rem = l - f * FRAME;
  int hh = rem / 40;
  int ww = rem - hh * 40;
#pragma unroll
  for (int j = 0; j < 3; j++) {
    int d = base + 2 * j;
    int doff = d & (HDIM - 1);
    int c = doff >> 1;
    int pos = (c < DT_C) ? f : ((c < DT_C + DS_C) ? hh : ww);
    float co = fc[pos * 64 + c];
    float si = fs[pos * 64 + c];
    float xr = v[2 * j] * rstd * gain[d];
    float xi = v[2 * j + 1] * rstd * gain[d + 1];
    float orr = xr * co - xi * si;
    float oii = xr * si + xi * co;
    int h = d >> 7;
    size_t oidx = ((size_t)h * L_TOK + l) * HDIM + doff;
    union { _Float16 h2[2]; unsigned u; } pk;
    pk.h2[0] = (_Float16)orr;
    pk.h2[1] = (_Float16)oii;
    *(unsigned*)(out + oidx) = pk.u;
  }
}

// ---------------- V fp32 [L, DIM] -> f16 transposed [NH][HD][L] ----------------
__global__ void vtrans_kernel(const float* __restrict__ V0, _Float16* __restrict__ Vt) {
  size_t i = (size_t)blockIdx.x * blockDim.x + threadIdx.x;
  if (i >= (size_t)L_TOK * DIMN) return;
  int l = (int)(i % L_TOK);
  int d = (int)(i / L_TOK);
  Vt[i] = (_Float16)V0[(size_t)l * DIMN + d];
}

// ---------------- attention: one wave per (16-query tile, head) ----------------
// S^T = K_tile * Q^T (queries in lanes), online softmax, O^T += V^T * P
__global__ __launch_bounds__(256) void attn_kernel(const _Float16* __restrict__ Qh,
                                                   const _Float16* __restrict__ Kh,
                                                   const _Float16* __restrict__ Vt,
                                                   _Float16* __restrict__ Oh) {
  int lane = threadIdx.x & 31;
  int wid = blockIdx.x * 8 + (threadIdx.x >> 5);
  int qt = wid % NQT;
  int h = wid / NQT;
  int q0 = qt * 16;
  int chunk = q0 / BLOCK_Q;
  int chunk_end = (chunk + 1) * BLOCK_Q;
  int rs[2], re[2];
  if (chunk == 0) { rs[0] = 0; re[0] = BLOCK_Q; rs[1] = 0; re[1] = 0; }
  else { rs[0] = 0; re[0] = SINK; rs[1] = chunk_end - BLOCK_Q; re[1] = chunk_end; }

  const _Float16* Qb = Qh + (size_t)h * L_TOK * HDIM;
  const _Float16* Kb = Kh + (size_t)h * L_TOK * HDIM;
  const _Float16* Vb = Vt + (size_t)h * HDIM * L_TOK;

  v16h qf[4];
#pragma unroll
  for (int kk = 0; kk < 4; kk++) qf[kk] = load_fragB(Qb, HDIM, q0, kk * 32, lane);

  v8f o[8];
  v8f zero = {};
#pragma unroll
  for (int t = 0; t < 8; t++) o[t] = zero;
  float run_max = -3.0e38f, run_sum = 0.f;
  const float sc = 0.08838834764831845f;  // 1/sqrt(128)

  for (int ri = 0; ri < 2; ri++) {
    for (int kb = rs[ri]; kb < re[ri]; kb += 32) {
      bool full = (kb + 32 <= re[ri]);  // wave-uniform
      v8f s0 = zero, s1 = zero;
#pragma unroll
      for (int kk = 0; kk < 4; kk++) {
        v16h ka = load_fragA(Kb, HDIM, kb, kk * 32, lane);
        s0 = wmma_f16(ka, qf[kk], s0);
      }
      if (full) {
#pragma unroll
        for (int kk = 0; kk < 4; kk++) {
          v16h ka = load_fragA(Kb, HDIM, kb + 16, kk * 32, lane);
          s1 = wmma_f16(ka, qf[kk], s1);
        }
      }
      float mloc = -3.0e38f;
#pragma unroll
      for (int r = 0; r < 8; r++) { s0[r] *= sc; mloc = fmaxf(mloc, s0[r]); }
      if (full) {
#pragma unroll
        for (int r = 0; r < 8; r++) { s1[r] *= sc; mloc = fmaxf(mloc, s1[r]); }
      }
      mloc = fmaxf(mloc, __shfl_xor(mloc, 16, 32));
      float nm = fmaxf(run_max, mloc);
      float alpha = __expf(run_max - nm);
      float lsum = 0.f;
      union { v8h h; unsigned u[4]; } p0, p1;
#pragma unroll
      for (int r = 0; r < 8; r++) {
        float p = __expf(s0[r] - nm);
        lsum += p;
        p0.h[r] = (_Float16)p;
      }
      if (full) {
#pragma unroll
        for (int r = 0; r < 8; r++) {
          float p = __expf(s1[r] - nm);
          lsum += p;
          p1.h[r] = (_Float16)p;
        }
      } else {
#pragma unroll
        for (int r = 0; r < 8; r++) p1.h[r] = (_Float16)0.f;
      }
      lsum += __shfl_xor(lsum, 16, 32);
      run_sum = run_sum * alpha + lsum;
      run_max = nm;

      // Assemble P as a B-fragment (32 keys x 16 queries).
      unsigned p0p[4], p1p[4];
#pragma unroll
      for (int i = 0; i < 4; i++) {
        p0p[i] = __shfl_xor(p0.u[i], 16, 32);
        p1p[i] = __shfl_xor(p1.u[i], 16, 32);
      }
      union { v16h h; unsigned u[8]; } pb;
      bool lowg = (lane < 16);
#pragma unroll
      for (int i = 0; i < 4; i++) {
        pb.u[i]     = lowg ? p0.u[i] : p1p[i];
        pb.u[4 + i] = lowg ? p0p[i] : p1.u[i];
      }
#pragma unroll
      for (int t = 0; t < 8; t++) {
#pragma unroll
        for (int r = 0; r < 8; r++) o[t][r] *= alpha;
      }
#pragma unroll
      for (int t = 0; t < 8; t++) {
        v16h av = load_fragA(Vb, L_TOK, t * 16, kb, lane);
        o[t] = wmma_f16(av, pb.h, o[t]);
      }
    }
  }
  float inv = 1.0f / run_sum;
  int n = lane & 15, g = lane >> 4;
  size_t rowbase = ((size_t)(q0 + n)) * DIMN + (size_t)h * HDIM;
#pragma unroll
  for (int t = 0; t < 8; t++) {
    v8h pk;
#pragma unroll
    for (int r = 0; r < 8; r++) pk[r] = (_Float16)(o[t][r] * inv);
    *(v8h*)(Oh + rowbase + t * 16 + g * 8) = pk;
  }
}

// ---------------- host orchestration ----------------
extern "C" void kernel_launch(void* const* d_in, const int* in_sizes, int n_in,
                              void* d_out, int out_size, void* d_ws, size_t ws_size,
                              hipStream_t stream) {
  const float* x  = (const float*)d_in[0];
  const float* Wq = (const float*)d_in[1];
  const float* bq = (const float*)d_in[2];
  const float* Wk = (const float*)d_in[3];
  const float* bk = (const float*)d_in[4];
  const float* Wv = (const float*)d_in[5];
  const float* bv = (const float*)d_in[6];
  const float* Wo = (const float*)d_in[7];
  const float* bo = (const float*)d_in[8];
  const float* gq = (const float*)d_in[9];
  const float* gk = (const float*)d_in[10];
  const float* fc = (const float*)d_in[11];
  const float* fs = (const float*)d_in[12];
  float* out = (float*)d_out;

  char* ws = (char*)d_ws;
  size_t off = 0;
  auto alloc = [&](size_t bytes) -> char* {
    char* p = ws + off;
    off += (bytes + 255) & ~(size_t)255;
    return p;
  };
  _Float16* Xh  = (_Float16*)alloc((size_t)L_TOK * DIMN * 2);
  _Float16* Wqh = (_Float16*)alloc((size_t)DIMN * DIMN * 2);
  _Float16* Wkh = (_Float16*)alloc((size_t)DIMN * DIMN * 2);
  _Float16* Wvh = (_Float16*)alloc((size_t)DIMN * DIMN * 2);
  _Float16* Woh = (_Float16*)alloc((size_t)DIMN * DIMN * 2);
  float*    T32 = (float*)alloc((size_t)L_TOK * DIMN * 4);
  _Float16* Qh  = (_Float16*)alloc((size_t)L_TOK * DIMN * 2);
  _Float16* Kh  = (_Float16*)alloc((size_t)L_TOK * DIMN * 2);
  _Float16* Vt  = (_Float16*)alloc((size_t)L_TOK * DIMN * 2);
  _Float16* Oh  = (_Float16*)alloc((size_t)L_TOK * DIMN * 2);

  long nx = (long)L_TOK * DIMN;
  long nw = (long)DIMN * DIMN;
  cast_kernel<<<(int)((nx + 255) / 256), 256, 0, stream>>>(x, Xh, nx);
  cast_kernel<<<(int)((nw + 255) / 256), 256, 0, stream>>>(Wq, Wqh, nw);
  cast_kernel<<<(int)((nw + 255) / 256), 256, 0, stream>>>(Wk, Wkh, nw);
  cast_kernel<<<(int)((nw + 255) / 256), 256, 0, stream>>>(Wv, Wvh, nw);
  cast_kernel<<<(int)((nw + 255) / 256), 256, 0, stream>>>(Wo, Woh, nw);

  dim3 ggrid(L_TOK / 128, DIMN / 128);

  gemm_kernel<<<ggrid, 256, 0, stream>>>(Xh, Wqh, bq, T32, DIMN, KDIM);
  rmsrope_kernel<<<L_TOK, 256, 0, stream>>>(T32, gq, fc, fs, Qh);

  gemm_kernel<<<ggrid, 256, 0, stream>>>(Xh, Wkh, bk, T32, DIMN, KDIM);
  rmsrope_kernel<<<L_TOK, 256, 0, stream>>>(T32, gk, fc, fs, Kh);

  gemm_kernel<<<ggrid, 256, 0, stream>>>(Xh, Wvh, bv, T32, DIMN, KDIM);
  vtrans_kernel<<<(int)((nx + 255) / 256), 256, 0, stream>>>(T32, Vt);

  attn_kernel<<<(NQT * NHEAD) / 8, 256, 0, stream>>>(Qh, Kh, Vt, Oh);

  gemm_kernel<<<ggrid, 256, 0, stream>>>(Oh, Woh, bo, out, DIMN, KDIM);
}
